// AttentionBlock_51049981280301
// MI455X (gfx1250) — matrix operand order repaired and checked by reading, not compiled, not run on hardware
//
#include <hip/hip_runtime.h>
#include <hip/hip_bf16.h>
#include <math.h>

typedef __attribute__((ext_vector_type(16))) _Float16 v16h;
typedef __attribute__((ext_vector_type(8)))  _Float16 v8h;
typedef __attribute__((ext_vector_type(8)))  float    v8f;
typedef _Float16 h16;

#define DEV static __device__ __forceinline__

constexpr int Bsz = 16, C = 512, HW = 1024, NH = 4, DH = 128, G = 32, CPG = 16;
constexpr float EPS = 1e-5f;

// ---- WMMA fragment builders (layouts per CDNA5 ISA 7.12.2) ----
// A fragment: 16x32 f16, source row-major with K contiguous. tile = &A[row0][k0]
DEV v16h frag_a_rm(const h16* tile, int ldk, int lane) {
  int row = lane & 15, hi = lane >> 4;
  const h16* p = tile + (size_t)row * ldk + 8 * hi;
  v8h lo = *(const v8h*)p;        // K = 8*hi + 0..7
  v8h hv = *(const v8h*)(p + 16); // K = 16 + 8*hi + 0..7
  v16h r;
#pragma unroll
  for (int e = 0; e < 8; ++e) { r[e] = lo[e]; r[8 + e] = hv[e]; }
  return r;
}

// B fragment: 32x16 f16, source = B^T stored row-major ([N][K]). tileT = &B^T[col0][k0]
DEV v16h frag_b_rm(const h16* tileT, int ldk, int lane) {
  int col = lane & 15, hi = lane >> 4;
  const h16* p = tileT + (size_t)col * ldk + 8 * hi; // K = 8*hi + 0..7, 16 + 8*hi + 0..7
  v8h a = *(const v8h*)p;
  v8h b = *(const v8h*)(p + 16);
  v16h r;
#pragma unroll
  for (int e = 0; e < 8; ++e) { r[e] = a[e]; r[8 + e] = b[e]; }
  return r;
}

DEV v8f wmma16(v16h a, v16h b, v8f c) {
  return __builtin_amdgcn_wmma_f32_16x16x32_f16(false, a, false, b, (short)0, c,
                                                false, false);
}

// Double-buffered GEMM panel: 16 (M) x 64 (N) output, K = C = 512.
// arow = &A[o0][0] (row-major, ldk=C), brow = &B^T[n0][0] (row-major, ldk=C).
// Prefetches the next k-step's fragments into fresh registers so the WMMAs of
// step k overlap the global loads of step k+32 (avoids s_wait_loadcnt 0 stalls).
DEV void gemm_16x64_kC(const h16* __restrict__ arow, const h16* __restrict__ brow,
                       int lane, v8f acc[4]) {
  v16h a = frag_a_rm(arow, C, lane);
  v16h b0 = frag_b_rm(brow + 0 * 16 * C, C, lane);
  v16h b1 = frag_b_rm(brow + 1 * 16 * C, C, lane);
  v16h b2 = frag_b_rm(brow + 2 * 16 * C, C, lane);
  v16h b3 = frag_b_rm(brow + 3 * 16 * C, C, lane);
  for (int k0 = 0; k0 < C - 32; k0 += 32) {
    int k1 = k0 + 32;
    v16h an = frag_a_rm(arow + k1, C, lane);
    v16h n0 = frag_b_rm(brow + 0 * 16 * C + k1, C, lane);
    v16h n1 = frag_b_rm(brow + 1 * 16 * C + k1, C, lane);
    v16h n2 = frag_b_rm(brow + 2 * 16 * C + k1, C, lane);
    v16h n3 = frag_b_rm(brow + 3 * 16 * C + k1, C, lane);
    acc[0] = wmma16(a, b0, acc[0]);
    acc[1] = wmma16(a, b1, acc[1]);
    acc[2] = wmma16(a, b2, acc[2]);
    acc[3] = wmma16(a, b3, acc[3]);
    a = an; b0 = n0; b1 = n1; b2 = n2; b3 = n3;
  }
  acc[0] = wmma16(a, b0, acc[0]);
  acc[1] = wmma16(a, b1, acc[1]);
  acc[2] = wmma16(a, b2, acc[2]);
  acc[3] = wmma16(a, b3, acc[3]);
}

// ---- Kernel 1: GroupNorm -> h^T f16 in [b][n][c] (K-contiguous for B frags) ----
__global__ __launch_bounds__(256) void k_groupnorm(const float* __restrict__ x,
                                                   const float* __restrict__ nw,
                                                   const float* __restrict__ nb,
                                                   h16* __restrict__ ht) {
  int b = blockIdx.x >> 5, g = blockIdx.x & 31;
  int t = threadIdx.x;
  const float* xg = x + (size_t)(b * C + g * CPG) * HW;
  float s = 0.f, s2 = 0.f;
  for (int i = t; i < CPG * HW; i += 256) { float v = xg[i]; s += v; s2 += v * v; }
  __shared__ float r1[256], r2[256];
  r1[t] = s; r2[t] = s2; __syncthreads();
  for (int o = 128; o > 0; o >>= 1) {
    if (t < o) { r1[t] += r1[t + o]; r2[t] += r2[t + o]; }
    __syncthreads();
  }
  float inv = 1.f / (float)(CPG * HW);
  float mu = r1[0] * inv;
  float var = r2[0] * inv - mu * mu;
  float rstd = rsqrtf(var + EPS);
  int cl = t & 15, c = g * CPG + cl;
  float w = nw[c] * rstd, bb = nb[c] - mu * w;
  for (int n = t >> 4; n < HW; n += 16) {
    float v = xg[(size_t)cl * HW + n];
    ht[((size_t)b * HW + n) * C + c] = (h16)(v * w + bb);
  }
}

// ---- Kernel 1b: weights fp32 -> f16 ----
__global__ void k_cvt_w(const float* __restrict__ qkvw, const float* __restrict__ projw,
                        h16* __restrict__ qkvh, h16* __restrict__ projh) {
  int i = blockIdx.x * 256 + threadIdx.x;
  if (i < 3 * C * C) qkvh[i] = (h16)qkvw[i];
  if (i < C * C)     projh[i] = (h16)projw[i];
}

// ---- Kernel 2: QKV GEMM. qkv[o][n] = sum_c W[o][c]*h[c][n]; scatter q/k/v ----
__global__ __launch_bounds__(256) void k_qkv(const h16* __restrict__ wq,
                                             const h16* __restrict__ ht,
                                             h16* __restrict__ qws,
                                             h16* __restrict__ kws,
                                             h16* __restrict__ vws) {
  int t = blockIdx.x;
  int nblk = t & 1; t >>= 1;
  int mt = t % 96, b = t / 96;
  int lane = threadIdx.x & 31, w = threadIdx.x >> 5;
  int o0 = mt * 16;
  int n0 = (nblk * 8 + w) * 64;
  v8f acc[4] = {};
  gemm_16x64_kC(wq + (size_t)o0 * C, ht + ((size_t)b * HW + n0) * C, lane, acc);

  int col = lane & 15, hi = lane >> 4;
#pragma unroll
  for (int j = 0; j < 4; ++j) {
    int n = n0 + j * 16 + col;
#pragma unroll
    for (int r = 0; r < 8; ++r) {
      int o = o0 + r + 8 * hi;
      h16 val = (h16)acc[j][r];
      int which = o >> 9, oc = o & 511;
      int hh = oc >> 7, dk = oc & 127;
      size_t qk = (((size_t)b * NH + hh) * HW + n) * DH + dk; // [b][h][n][d]
      if (which == 0)      qws[qk] = val;
      else if (which == 1) kws[qk] = val;
      else vws[(((size_t)b * NH + hh) * DH + dk) * HW + n] = val; // [b][h][d][n]
    }
  }
}

// ---- Kernel 3: attention for one (b, head, 16-row tile). 4 waves. ----
__global__ __launch_bounds__(128) void k_attn(const h16* __restrict__ qws,
                                              const h16* __restrict__ kws,
                                              const h16* __restrict__ vws,
                                              h16* __restrict__ ows) {
  __shared__ float sc[16][HW]; // 64 KB score tile
  int bi = blockIdx.x;
  int ntile = bi & 63, hh = (bi >> 6) & 3, b = bi >> 8;
  int n0 = ntile * 16;
  int lane = threadIdx.x & 31, w = threadIdx.x >> 5;
  int col = lane & 15, hi = lane >> 4;
  const h16* qb = qws + (((size_t)b * NH + hh) * HW + n0) * DH;
  const h16* kb = kws + ((size_t)b * NH + hh) * HW * DH;
  const h16* vb = vws + ((size_t)b * NH + hh) * DH * HW;
  const float scale = 0.08838834764831845f; // 128^-0.5

  // Q fragments reused across all 64 m-tiles (K = 128 -> 4 steps)
  v16h aq[4];
#pragma unroll
  for (int ks = 0; ks < 4; ++ks) aq[ks] = frag_a_rm(qb + ks * 32, DH, lane);

  for (int mt = w; mt < 64; mt += 4) {
    const h16* kbase = kb + (size_t)mt * 16 * DH;
    v8f acc = {};
    v16h bk = frag_b_rm(kbase, DH, lane);
#pragma unroll
    for (int ks = 0; ks < 4; ++ks) {
      v16h bn;
      if (ks < 3) bn = frag_b_rm(kbase + (ks + 1) * 32, DH, lane); // prefetch
      acc = wmma16(aq[ks], bk, acc);
      bk = bn;
    }
#pragma unroll
    for (int r = 0; r < 8; ++r) sc[r + 8 * hi][mt * 16 + col] = acc[r] * scale;
  }
  __syncthreads();

  // softmax: 8 lanes per row (intra-wave), normalize in place
  {
    int row = threadIdx.x >> 3, sub = threadIdx.x & 7;
    float m = -3.4e38f;
    for (int j = 0; j < 128; ++j) m = fmaxf(m, sc[row][j * 8 + sub]);
#pragma unroll
    for (int off = 1; off < 8; off <<= 1) m = fmaxf(m, __shfl_xor(m, off, 32));
    float s = 0.f;
    for (int j = 0; j < 128; ++j) {
      float e = __expf(sc[row][j * 8 + sub] - m);
      sc[row][j * 8 + sub] = e;
      s += e;
    }
#pragma unroll
    for (int off = 1; off < 8; off <<= 1) s += __shfl_xor(s, off, 32);
    float is = 1.f / s;
    for (int j = 0; j < 128; ++j) sc[row][j * 8 + sub] *= is;
  }
  __syncthreads();

  // O[n][dk] = sum_m P[n][m] * v[dk][m]; each wave does 2 of 8 dk-tiles.
  // Double-buffer V fragments; A fragment built on the fly from f32 LDS probs.
  int dk0a = (w * 2 + 0) * 16, dk0b = (w * 2 + 1) * 16;
  const h16* va = vb + (size_t)dk0a * HW;
  const h16* vbp = vb + (size_t)dk0b * HW;
  v8f acc2[2] = {};
  v16h bva = frag_b_rm(va, HW, lane);
  v16h bvb = frag_b_rm(vbp, HW, lane);
  for (int k0 = 0; k0 < HW - 32; k0 += 32) {
    int k1 = k0 + 32;
    v16h na = frag_b_rm(va + k1, HW, lane);   // prefetch next V frags
    v16h nb2 = frag_b_rm(vbp + k1, HW, lane);
    v16h ap;
    {
      const float* rp = &sc[col][0] + k0 + 8 * hi;
#pragma unroll
      for (int e = 0; e < 8; ++e) { ap[e] = (h16)rp[e]; ap[8 + e] = (h16)rp[16 + e]; }
    }
    acc2[0] = wmma16(ap, bva, acc2[0]);
    acc2[1] = wmma16(ap, bvb, acc2[1]);
    bva = na; bvb = nb2;
  }
  {
    v16h ap;
    const float* rp = &sc[col][0] + (HW - 32) + 8 * hi;
#pragma unroll
    for (int e = 0; e < 8; ++e) { ap[e] = (h16)rp[e]; ap[8 + e] = (h16)rp[16 + e]; }
    acc2[0] = wmma16(ap, bva, acc2[0]);
    acc2[1] = wmma16(ap, bvb, acc2[1]);
  }
#pragma unroll
  for (int dt = 0; dt < 2; ++dt) {
    int c = hh * DH + (w * 2 + dt) * 16 + col;
#pragma unroll
    for (int r = 0; r < 8; ++r) {
      int n = n0 + r + 8 * hi;
      ows[((size_t)b * HW + n) * C + c] = (h16)acc2[dt][r]; // o^T [b][n][c]
    }
  }
}

// ---- Kernel 4: proj GEMM + bias + residual (f32 out) ----
__global__ __launch_bounds__(256) void k_proj(const h16* __restrict__ pw,
                                              const h16* __restrict__ ows,
                                              const float* __restrict__ x,
                                              const float* __restrict__ pb,
                                              float* __restrict__ out) {
  int t = blockIdx.x;
  int nblk = t & 1; t >>= 1;
  int mt = t & 31, b = t >> 5;
  int lane = threadIdx.x & 31, w = threadIdx.x >> 5;
  int o0 = mt * 16;
  int n0 = (nblk * 8 + w) * 64;
  v8f acc[4] = {};
  gemm_16x64_kC(pw + (size_t)o0 * C, ows + ((size_t)b * HW + n0) * C, lane, acc);

  int col = lane & 15, hi = lane >> 4;
#pragma unroll
  for (int j = 0; j < 4; ++j) {
    int n = n0 + j * 16 + col;
#pragma unroll
    for (int r = 0; r < 8; ++r) {
      int co = o0 + r + 8 * hi;
      size_t idx = ((size_t)b * C + co) * HW + n;
      out[idx] = x[idx] + pb[co] + acc[j][r];
    }
  }
}

extern "C" void kernel_launch(void* const* d_in, const int* in_sizes, int n_in,
                              void* d_out, int out_size, void* d_ws, size_t ws_size,
                              hipStream_t stream) {
  const float* x     = (const float*)d_in[0];
  const float* nw    = (const float*)d_in[1];
  const float* nb    = (const float*)d_in[2];
  const float* qkvw  = (const float*)d_in[3];
  const float* projw = (const float*)d_in[4];
  const float* pb    = (const float*)d_in[5];
  float* out = (float*)d_out;

  h16* ws = (h16*)d_ws;
  size_t off = 0;
  h16* ht    = ws + off; off += (size_t)Bsz * HW * C;       // h^T  [b][n][c]
  h16* qkvh  = ws + off; off += (size_t)3 * C * C;          // qkv_w f16
  h16* projh = ws + off; off += (size_t)C * C;              // proj_w f16
  h16* qws   = ws + off; off += (size_t)Bsz * NH * HW * DH; // q [b][h][n][d]
  h16* kws   = ws + off; off += (size_t)Bsz * NH * HW * DH; // k [b][h][n][d]
  h16* vws   = ws + off; off += (size_t)Bsz * NH * DH * HW; // v [b][h][d][n]
  h16* ows   = ws + off;                                    // o^T [b][n][c]

  k_groupnorm<<<Bsz * G, 256, 0, stream>>>(x, nw, nb, ht);
  k_cvt_w<<<(3 * C * C + 255) / 256, 256, 0, stream>>>(qkvw, projw, qkvh, projh);
  k_qkv<<<Bsz * 96 * 2, 256, 0, stream>>>(qkvh, ht, qws, kws, vws);
  k_attn<<<Bsz * NH * 64, 128, 0, stream>>>(qws, kws, vws, ows);
  k_proj<<<Bsz * 32 * 2, 256, 0, stream>>>(projh, ows, x, pb, out);
}
